// BiLSTM_CRF_21766894256599
// MI455X (gfx1250) — matrix operand order repaired and checked by reading, not compiled, not run on hardware
//
#include <hip/hip_runtime.h>
#include <hip/hip_bf16.h>
#include <math.h>

// ---------------------------------------------------------------------------
// BiLSTM-CRF forward loss for MI455X (gfx1250, wave32, WMMA).
// Phases:
//   1) build x_bf16 [T*B, 544]  (emb gather + eeg[:, :, 4:6] concat, zero-pad)
//   2) convert w_ih (pad 514->544) and w_hh to bf16
//   3) pre-gate GEMM (bf16 WMMA, f32 accum):  pre = x @ w_ih^T + b   [T,B,2048]
//      register-blocked 2 Mtiles x 4 Ntiles per wave, launch_bounds(256)
//      so the 8 accumulator tiles stay in VGPRs (no scratch spills)
//   4) persistent LSTM scan (2 blocks = 2 directions, 32 waves each,
//      h double-buffered in LDS, cell state in VGPRs, bf16 WMMA per step,
//      global_prefetch of next step's pre-gates)
//   5) layernorm + relu + emissions (K=6)    [T,B,6]
//   6) CRF numerator/forward-algorithm scan, scalar -mean(ll)
// ---------------------------------------------------------------------------

#define BB 32      // batch
#define TT 512     // time
#define HH 512     // hidden
#define KK 6       // tags
#define DIN 514    // D + 2
#define DINP 544   // padded to multiple of 32
#define G4 2048    // 4*H

typedef __attribute__((ext_vector_type(8)))  __bf16 v8bf;
typedef __attribute__((ext_vector_type(16))) __bf16 v16bf;
typedef __attribute__((ext_vector_type(8)))  float  v8f;

__device__ __forceinline__ v8f wmma_bf16(v16bf a, v16bf b, v8f c) {
  return __builtin_amdgcn_wmma_f32_16x16x32_bf16(false, a, false, b, (short)0, c,
                                                 false, false);
}

// A operand: 16x32 bf16, M rows of `base` (row-major, stride elems).
// lanes 0-15: M=lane,  K = kbase+{0..15}
// lanes16-31: M=lane-16, K = kbase+{16..31}
__device__ __forceinline__ v16bf load_A16x32(const __bf16* base, int stride,
                                             int mbase, int kbase, int lane) {
  int m    = mbase + (lane & 15);
  int koff = kbase + (lane & 16);  // +16 for upper half
  const __bf16* p = base + (size_t)m * stride + koff;
  v8bf lo = *(const v8bf*)p;
  v8bf hi = *(const v8bf*)(p + 8);
  v16bf a;
#pragma unroll
  for (int i = 0; i < 8; ++i) { a[i] = lo[i]; a[i + 8] = hi[i]; }
  return a;
}

// B operand: 32x16 (KxN) where w is row-major [N_total, K_total] (w[g,k]).
// lane holds N = nbase + lane%16, 16 contiguous K starting kbase (+16 upper).
__device__ __forceinline__ v16bf load_B32x16(const __bf16* w, int stride,
                                             int nbase, int kbase, int lane) {
  int g    = nbase + (lane & 15);
  int koff = kbase + (lane & 16);
  return *(const v16bf*)(w + (size_t)g * stride + koff);
}

__device__ __forceinline__ float wave_sum(float v) {
#pragma unroll
  for (int m = 16; m > 0; m >>= 1) v += __shfl_xor(v, m, 32);
  return v;
}

__device__ __forceinline__ float sigmoidf(float x) {
  return 1.0f / (1.0f + expf(-x));
}

// ------------------------- phase 1: build x ---------------------------------
__global__ void k_build_x(const int* __restrict__ ids,
                          const float* __restrict__ eeg,
                          const float* __restrict__ emb,
                          __bf16* __restrict__ x) {
  int idx = blockIdx.x * blockDim.x + threadIdx.x;
  if (idx >= TT * BB * DINP) return;
  int row = idx / DINP;        // t*B + b
  int col = idx - row * DINP;
  int t = row >> 5, b = row & 31;
  float v = 0.0f;
  if (col < 512) {
    int id = ids[b * TT + t];
    v = emb[(size_t)id * 512 + col];
  } else if (col < DIN) {
    v = eeg[((size_t)b * TT + t) * 8 + 4 + (col - 512)];
  }
  x[idx] = (__bf16)v;
}

// ------------------------- phase 2: weight convert --------------------------
__global__ void k_cvt_pad(const float* __restrict__ src, __bf16* __restrict__ dst,
                          int rows, int scols, int dcols) {
  int idx = blockIdx.x * blockDim.x + threadIdx.x;
  if (idx >= rows * dcols) return;
  int r = idx / dcols, c = idx - r * dcols;
  dst[idx] = (c < scols) ? (__bf16)src[(size_t)r * scols + c] : (__bf16)0.0f;
}

// ------------------------- phase 3: pre-gate GEMM ---------------------------
// pre[row, g] = sum_k x[row,k]*w[g,k] + bias[g], rows = T*B, g in [0,2048).
// One wave per (2 mtiles, 4 ntiles): A reused 4x, B reused 2x. 17 k-chunks.
// launch_bounds(256): 8 waves/WGP -> full VGPR budget, keeps 8 acc tiles
// (64 VGPRs) + operands resident with no scratch spills.
__global__ void __launch_bounds__(256, 1)
k_pregemm(const __bf16* __restrict__ x,
          const __bf16* __restrict__ w,
          const float* __restrict__ bias,
          float* __restrict__ pre) {
  int lane = threadIdx.x & 31;
  int gid  = (blockIdx.x * blockDim.x + threadIdx.x) >> 5;  // wave id 0..16383
  int mp   = gid >> 5;         // mtile pair 0..511
  int ng   = gid & 31;         // 0..31
  int n0   = ng * 64;          // 4 tiles * 16 cols
  int m0   = mp * 32;
  v8f acc[2][4];
#pragma unroll
  for (int r = 0; r < 2; ++r)
#pragma unroll
    for (int i = 0; i < 4; ++i) acc[r][i] = (v8f)0.0f;

  for (int kb = 0; kb < DINP; kb += 32) {
    v16bf a0 = load_A16x32(x, DINP, m0,      kb, lane);
    v16bf a1 = load_A16x32(x, DINP, m0 + 16, kb, lane);
#pragma unroll
    for (int tn = 0; tn < 4; ++tn) {
      v16bf bm = load_B32x16(w, DINP, n0 + tn * 16, kb, lane);
      acc[0][tn] = wmma_bf16(a0, bm, acc[0][tn]);
      acc[1][tn] = wmma_bf16(a1, bm, acc[1][tn]);
    }
  }
  int nlane = lane & 15;
  int moff  = (lane & 16) ? 8 : 0;
#pragma unroll
  for (int tn = 0; tn < 4; ++tn) {
    int col = n0 + tn * 16 + nlane;
    float bv = bias[col];
#pragma unroll
    for (int r = 0; r < 2; ++r) {
#pragma unroll
      for (int v = 0; v < 8; ++v) {
        int row = m0 + r * 16 + v + moff;
        pre[(size_t)row * G4 + col] = acc[r][tn][v] + bv;
      }
    }
  }
}

// ------------------------- phase 4: LSTM scan -------------------------------
// blockIdx.x = 0 forward, 1 backward. 1024 threads = 32 waves.
// Wave w owns H-columns [16w,16w+16) for all 4 gates -> cell update lane-local.
__global__ void __launch_bounds__(1024, 1)
k_lstm(const float* __restrict__ preF, const float* __restrict__ preB,
       const __bf16* __restrict__ whhF, const __bf16* __restrict__ whhB,
       float* __restrict__ hfo, float* __restrict__ hbo) {
  const bool rev = (blockIdx.x == 1);
  const float*  pre = rev ? preB : preF;
  const __bf16* whh = rev ? whhB : whhF;
  float*        hs  = rev ? hbo : hfo;

  __shared__ __bf16 hbuf[2][BB * HH];  // 64 KB double buffer
  for (int i = threadIdx.x; i < BB * HH; i += 1024) hbuf[0][i] = (__bf16)0.0f;
  __syncthreads();

  int lane = threadIdx.x & 31;
  int w    = threadIdx.x >> 5;          // wave/column group 0..31
  int c    = w * 16 + (lane & 15);      // owned hidden column (N)
  int moff = (lane & 16) ? 8 : 0;

  float cst[16];
#pragma unroll
  for (int i = 0; i < 16; ++i) cst[i] = 0.0f;

  int buf = 0;
  for (int step = 0; step < TT; ++step) {
    int t = rev ? (TT - 1 - step) : step;

    // Prefetch next step's pre-gate values (hides L2 latency behind WMMAs).
    if (step + 1 < TT) {
      int tn = rev ? (TT - 2 - step) : (step + 1);
      const float* pn = pre + (size_t)tn * BB * G4;
#pragma unroll
      for (int q = 0; q < 4; ++q)
        __builtin_prefetch(pn + (size_t)moff * G4 + q * HH + c, 0, 0);
    }

    v8f acc[4][2];
#pragma unroll
    for (int q = 0; q < 4; ++q) { acc[q][0] = (v8f)0.0f; acc[q][1] = (v8f)0.0f; }

    const __bf16* hsrc = &hbuf[buf][0];
    for (int kb = 0; kb < HH; kb += 32) {
      v16bf a0 = load_A16x32(hsrc, HH, 0,  kb, lane);
      v16bf a1 = load_A16x32(hsrc, HH, 16, kb, lane);
#pragma unroll
      for (int q = 0; q < 4; ++q) {
        v16bf bm = load_B32x16(whh, HH, q * HH + w * 16, kb, lane);
        acc[q][0] = wmma_bf16(a0, bm, acc[q][0]);
        acc[q][1] = wmma_bf16(a1, bm, acc[q][1]);
      }
    }

    const float* pre_t = pre + (size_t)t * BB * G4;
    float*       hs_t  = hs  + (size_t)t * BB * HH;
    int nbuf = buf ^ 1;
#pragma unroll
    for (int rt = 0; rt < 2; ++rt) {
#pragma unroll
      for (int v = 0; v < 8; ++v) {
        int m = rt * 16 + v + moff;  // batch row
        const float* pm = pre_t + (size_t)m * G4 + c;
        float gi = acc[0][rt][v] + pm[0 * HH];
        float gf = acc[1][rt][v] + pm[1 * HH];
        float gg = acc[2][rt][v] + pm[2 * HH];
        float go = acc[3][rt][v] + pm[3 * HH];
        float cc = sigmoidf(gf) * cst[rt * 8 + v] + sigmoidf(gi) * tanhf(gg);
        cst[rt * 8 + v] = cc;
        float hh = sigmoidf(go) * tanhf(cc);
        hbuf[nbuf][m * HH + c] = (__bf16)hh;
        hs_t[(size_t)m * HH + c] = hh;
      }
    }
    __syncthreads();
    buf = nbuf;
  }
}

// ------------------- phase 5: layernorm + relu + emissions ------------------
__global__ void __launch_bounds__(256)
k_ln_em(const float* __restrict__ hf, const float* __restrict__ hb,
        const float* __restrict__ ln_g, const float* __restrict__ ln_b,
        const float* __restrict__ w_out, const float* __restrict__ b_out,
        float* __restrict__ em) {
  int lane = threadIdx.x & 31;
  int item = blockIdx.x * 8 + (threadIdx.x >> 5);  // 0..16383 = t*32+b
  const float* pf = hf + (size_t)item * HH;
  const float* pb = hb + (size_t)item * HH;

  float vals[32];
  float s = 0.0f;
#pragma unroll
  for (int j = 0; j < 32; ++j) {
    int e = lane + 32 * j;
    float v = (e < HH) ? pf[e] : pb[e - HH];
    vals[j] = v; s += v;
  }
  float mu = wave_sum(s) * (1.0f / 1024.0f);
  float vs = 0.0f;
#pragma unroll
  for (int j = 0; j < 32; ++j) { float d = vals[j] - mu; vs += d * d; }
  float var  = wave_sum(vs) * (1.0f / 1024.0f);
  float rinv = rsqrtf(var + 1e-5f);

  float dot[KK];
#pragma unroll
  for (int k = 0; k < KK; ++k) dot[k] = 0.0f;
#pragma unroll
  for (int j = 0; j < 32; ++j) {
    int e = lane + 32 * j;
    float n = (vals[j] - mu) * rinv * ln_g[e] + ln_b[e];
    n = fmaxf(n, 0.0f);
#pragma unroll
    for (int k = 0; k < KK; ++k) dot[k] += n * w_out[k * 1024 + e];
  }
#pragma unroll
  for (int k = 0; k < KK; ++k) dot[k] = wave_sum(dot[k]);
  if (lane == 0) {
#pragma unroll
    for (int k = 0; k < KK; ++k) em[(size_t)item * KK + k] = dot[k] + b_out[k];
  }
}

// ------------------------- phase 6: CRF scan --------------------------------
__global__ void k_crf(const float* __restrict__ em, const int* __restrict__ tags,
                      const int* __restrict__ amask,
                      const float* __restrict__ start_t, const float* __restrict__ end_t,
                      const float* __restrict__ trans, float* __restrict__ out) {
  int b = threadIdx.x;  // one lane per batch element
  float tr[KK * KK], st[KK], en[KK];
#pragma unroll
  for (int i = 0; i < KK * KK; ++i) tr[i] = trans[i];
#pragma unroll
  for (int i = 0; i < KK; ++i) { st[i] = start_t[i]; en[i] = end_t[i]; }

  int tag0 = tags[b * TT];
  float e0[KK];
#pragma unroll
  for (int k = 0; k < KK; ++k) e0[k] = em[(size_t)b * KK + k];
  float score = st[tag0] + e0[tag0];
  float alpha[KK];
#pragma unroll
  for (int k = 0; k < KK; ++k) alpha[k] = st[k] + e0[k];
  int prev = tag0;
  float msum = (amask[b * TT] != 0) ? 1.0f : 0.0f;

  for (int t = 1; t < TT; ++t) {
    float mt = (amask[b * TT + t] != 0) ? 1.0f : 0.0f;
    msum += mt;
    int tg = tags[b * TT + t];
    float e[KK];
#pragma unroll
    for (int k = 0; k < KK; ++k) e[k] = em[((size_t)t * BB + b) * KK + k];
    score += mt * (tr[prev * KK + tg] + e[tg]);
    float nxt[KK];
#pragma unroll
    for (int j = 0; j < KK; ++j) {
      float mx = -1e30f;
#pragma unroll
      for (int i = 0; i < KK; ++i) mx = fmaxf(mx, alpha[i] + tr[i * KK + j]);
      float sm = 0.0f;
#pragma unroll
      for (int i = 0; i < KK; ++i) sm += expf(alpha[i] + tr[i * KK + j] - mx);
      nxt[j] = mx + logf(sm) + e[j];
    }
    if (mt > 0.0f) {
#pragma unroll
      for (int j = 0; j < KK; ++j) alpha[j] = nxt[j];
    }
    prev = tg;
  }
  int seq_end = (int)msum - 1;
  int last = tags[b * TT + seq_end];
  float num = score + en[last];
  float mx = -1e30f;
#pragma unroll
  for (int j = 0; j < KK; ++j) mx = fmaxf(mx, alpha[j] + en[j]);
  float sm = 0.0f;
#pragma unroll
  for (int j = 0; j < KK; ++j) sm += expf(alpha[j] + en[j] - mx);
  float ll = num - (mx + logf(sm));
  ll = wave_sum(ll);
  if (b == 0) out[0] = -(ll / (float)BB);
}

// ---------------------------------------------------------------------------
extern "C" void kernel_launch(void* const* d_in, const int* in_sizes, int n_in,
                              void* d_out, int out_size, void* d_ws, size_t ws_size,
                              hipStream_t stream) {
  const int*   input_ids = (const int*)d_in[0];
  const float* eeg       = (const float*)d_in[1];
  const int*   tags      = (const int*)d_in[2];
  const int*   amask     = (const int*)d_in[3];   // bool mask delivered as int
  const float* emb       = (const float*)d_in[4];
  const float* w_ih_f    = (const float*)d_in[5];
  const float* w_hh_f    = (const float*)d_in[6];
  const float* b_f       = (const float*)d_in[7];
  const float* w_ih_b    = (const float*)d_in[8];
  const float* w_hh_b    = (const float*)d_in[9];
  const float* b_b       = (const float*)d_in[10];
  const float* ln_g      = (const float*)d_in[11];
  const float* ln_b      = (const float*)d_in[12];
  const float* w_out     = (const float*)d_in[13];
  const float* b_out     = (const float*)d_in[14];
  const float* start_t   = (const float*)d_in[15];
  const float* end_t     = (const float*)d_in[16];
  const float* trans     = (const float*)d_in[17];

  // workspace layout (all chunk sizes are multiples of 256 bytes)
  char* ws = (char*)d_ws;
  size_t off = 0;
  __bf16* x_bf   = (__bf16*)(ws + off); off += (size_t)TT * BB * DINP * 2;
  __bf16* wihF   = (__bf16*)(ws + off); off += (size_t)G4 * DINP * 2;
  __bf16* wihB   = (__bf16*)(ws + off); off += (size_t)G4 * DINP * 2;
  __bf16* whhF   = (__bf16*)(ws + off); off += (size_t)G4 * HH * 2;
  __bf16* whhB   = (__bf16*)(ws + off); off += (size_t)G4 * HH * 2;
  float*  preF   = (float*)(ws + off);  off += (size_t)TT * BB * G4 * 4;
  float*  preB   = (float*)(ws + off);  off += (size_t)TT * BB * G4 * 4;
  float*  hf     = (float*)(ws + off);  off += (size_t)TT * BB * HH * 4;
  float*  hb     = (float*)(ws + off);  off += (size_t)TT * BB * HH * 4;
  float*  em     = (float*)(ws + off);  off += (size_t)TT * BB * KK * 4;

  // 1) build x
  {
    int n = TT * BB * DINP;
    k_build_x<<<(n + 255) / 256, 256, 0, stream>>>(input_ids, eeg, emb, x_bf);
  }
  // 2) weight conversion (pad Din 514 -> 544 for the ih matrices)
  {
    int n1 = G4 * DINP, n2 = G4 * HH;
    k_cvt_pad<<<(n1 + 255) / 256, 256, 0, stream>>>(w_ih_f, wihF, G4, DIN, DINP);
    k_cvt_pad<<<(n1 + 255) / 256, 256, 0, stream>>>(w_ih_b, wihB, G4, DIN, DINP);
    k_cvt_pad<<<(n2 + 255) / 256, 256, 0, stream>>>(w_hh_f, whhF, G4, HH, HH);
    k_cvt_pad<<<(n2 + 255) / 256, 256, 0, stream>>>(w_hh_b, whhB, G4, HH, HH);
  }
  // 3) pre-gate GEMMs: 512 mtile-pairs * 32 ngroups = 16384 waves / 8 per block
  {
    dim3 grid(2048), block(256);
    k_pregemm<<<grid, block, 0, stream>>>(x_bf, wihF, b_f, preF);
    k_pregemm<<<grid, block, 0, stream>>>(x_bf, wihB, b_b, preB);
  }
  // 4) LSTM scan (2 persistent workgroups, one per direction)
  k_lstm<<<2, 1024, 0, stream>>>(preF, preB, whhF, whhB, hf, hb);
  // 5) layernorm + relu + emissions: 16384 items, wave per item
  k_ln_em<<<2048, 256, 0, stream>>>(hf, hb, ln_g, ln_b, w_out, b_out, em);
  // 6) CRF scan -> scalar loss
  k_crf<<<1, 32, 0, stream>>>(em, tags, amask, start_t, end_t, trans, (float*)d_out);
}
